// VectorQuantizerEMA_26293789786235
// MI455X (gfx1250) — matrix operand order repaired and checked by reading, not compiled, not run on hardware
//
#include <hip/hip_runtime.h>
#include <hip/hip_bf16.h>
#include <math.h>

typedef __attribute__((ext_vector_type(16))) __bf16 v16bf;
typedef __attribute__((ext_vector_type(8)))  __bf16 v8bf;
typedef __attribute__((ext_vector_type(8)))  float  v8f;
typedef int async_v4i __attribute__((vector_size(16)));   // matches builtin param type

#define N_VEC   8192   // b*h*w = 8*32*32
#define K_CODES 8192
#define C_DIM   256
#define COL_PAD 272    // 256 + 16 bf16 pad -> 544B stride, spreads LDS banks

#if defined(__gfx1250__) && __has_builtin(__builtin_amdgcn_global_load_async_to_lds_b128)
#define USE_ASYNC 1
#define ASYNC_G(p) ((__attribute__((address_space(1))) async_v4i*)(p))
#define ASYNC_L(p) ((__attribute__((address_space(3))) async_v4i*)(p))
#else
#define USE_ASYNC 0
#endif

#if defined(__gfx1250__) && __has_builtin(__builtin_amdgcn_s_wait_asynccnt)
#define WAIT_ASYNC() __builtin_amdgcn_s_wait_asynccnt(0)
#else
#define WAIT_ASYNC() asm volatile("s_wait_asynccnt 0" ::: "memory")
#endif

// ---------------- prep: split embedding into bf16 hi/lo + row norms ----------
__global__ void prep_embed(const float* __restrict__ emb, __bf16* __restrict__ ehi,
                           __bf16* __restrict__ elo, float* __restrict__ enorm) {
    const int k = blockIdx.x, t = threadIdx.x;
    const size_t o = (size_t)k * C_DIM + t;
    float v = emb[o];
    __bf16 h = (__bf16)v;
    __bf16 l = (__bf16)(v - (float)h);
    ehi[o] = h;
    elo[o] = l;
    __shared__ float red[256];
    red[t] = v * v;
    __syncthreads();
    for (int s = 128; s > 0; s >>= 1) { if (t < s) red[t] += red[t + s]; __syncthreads(); }
    if (t == 0) enorm[k] = red[0];
}

// ---------------- prep: transpose inputs (b,c,h,w)->(n,c) and split ----------
__global__ void prep_inputs(const float* __restrict__ in, __bf16* __restrict__ xhi,
                            __bf16* __restrict__ xlo) {
    const int n = blockIdx.x, t = threadIdx.x;        // n = b*1024 + h*32 + w
    const int b = n >> 10, hw = n & 1023;
    float v = in[(size_t)b * 262144 + (size_t)t * 1024 + hw];
    __bf16 h = (__bf16)v;
    __bf16 l = (__bf16)(v - (float)h);
    const size_t o = (size_t)n * C_DIM + t;
    xhi[o] = h;
    xlo[o] = l;
}

// ---------------- core: WMMA distance GEMM + per-row argmin ------------------
// Block = 256 threads = 8 waves. 64 rows per block (4 row-blocks of 16).
// Wave w: row-block (w&3), column stripe (w>>2). Each 32-column codebook tile
// is staged once into double-buffered LDS and shared by 4 waves per stripe.
__global__ void __launch_bounds__(256)
vq_argmin(const __bf16* __restrict__ xhi, const __bf16* __restrict__ xlo,
          const __bf16* __restrict__ ehi, const __bf16* __restrict__ elo,
          const float* __restrict__ enorm, int* __restrict__ idxOut) {
    const int tid     = threadIdx.x;
    const int lane    = tid & 31;
    const int wave    = tid >> 5;
    const int rb      = wave & 3;        // row block
    const int ss      = wave >> 2;       // column stripe within tile
    const int rowBase = (blockIdx.x << 6) + (rb << 4);
    const int halfSel = lane >> 4;
    const int rowIn   = lane & 15;

    __shared__ __bf16 sB[2][2][2][16][COL_PAD];   // [buf][stripe][hi/lo][col][c]
    __shared__ float  sMin[8][32][8];
    __shared__ int    sIdx[8][32][8];

    // ---- A fragments (16 rows x 256 c, hi+lo) resident in registers --------
    v16bf aHi[8], aLo[8];
    {
        const __bf16* xrh = xhi + (size_t)(rowBase + rowIn) * C_DIM;
        const __bf16* xrl = xlo + (size_t)(rowBase + rowIn) * C_DIM;
#pragma unroll
        for (int f = 0; f < 8; ++f) {
            const int k0 = f * 32 + halfSel * 8;
            v8bf h0 = *(const v8bf*)(xrh + k0);
            v8bf h1 = *(const v8bf*)(xrh + k0 + 16);
            v8bf l0 = *(const v8bf*)(xrl + k0);
            v8bf l1 = *(const v8bf*)(xrl + k0 + 16);
#pragma unroll
            for (int e = 0; e < 8; ++e) {
                aHi[f][e] = h0[e]; aHi[f][8 + e] = h1[e];
                aLo[f][e] = l0[e]; aLo[f][8 + e] = l1[e];
            }
        }
    }

    float runMin[8]; int runIdx[8];
#pragma unroll
    for (int v = 0; v < 8; ++v) { runMin[v] = 3.0e38f; runIdx[v] = 0; }

    const int NIT = K_CODES / 32;        // 32 columns staged per iteration

    // Chunk decode: q in [0,2048): 16B chunks covering 2 stripes x hi/lo x 16 cols
#define CHUNK_SRC(q, colBase)                                                     \
    ((((q) >> 9) & 1 ? elo : ehi) +                                               \
     ((size_t)((colBase) + (((q) >> 10) << 4) + (((q) >> 5) & 15)) << 8) +        \
     (((q) & 31) << 3))
#define CHUNK_DST(q, buf)                                                         \
    (&sB[(buf)][((q) >> 10) & 1][((q) >> 9) & 1][((q) >> 5) & 15][((q) & 31) << 3])

#if USE_ASYNC
    // ---- async-to-LDS staging (gfx1250 ASYNCcnt path) ----------------------
    {
#pragma unroll
        for (int j = 0; j < 8; ++j) {
            const int q = j * 256 + tid;
            __builtin_amdgcn_global_load_async_to_lds_b128(
                ASYNC_G(CHUNK_SRC(q, 0)), ASYNC_L(CHUNK_DST(q, 0)), 0, 0);
        }
    }
    WAIT_ASYNC();
    __syncthreads();
#else
    // ---- fallback: register-staged copy ------------------------------------
    uint4 tr[8];
    {
#pragma unroll
        for (int j = 0; j < 8; ++j) {
            const int q = j * 256 + tid;
            tr[j] = *(const uint4*)CHUNK_SRC(q, 0);
        }
#pragma unroll
        for (int j = 0; j < 8; ++j) {
            const int q = j * 256 + tid;
            *(uint4*)CHUNK_DST(q, 0) = tr[j];
        }
    }
    __syncthreads();
#endif

    for (int it = 0; it < NIT; ++it) {
        const int buf = it & 1;
        const bool more = (it + 1) < NIT;
        const int nextCol = (it + 1) << 5;

#if USE_ASYNC
        if (more) {
#pragma unroll
            for (int j = 0; j < 8; ++j) {
                const int q = j * 256 + tid;
                __builtin_amdgcn_global_load_async_to_lds_b128(
                    ASYNC_G(CHUNK_SRC(q, nextCol)), ASYNC_L(CHUNK_DST(q, buf ^ 1)), 0, 0);
            }
        }
#else
        if (more) {
#pragma unroll
            for (int j = 0; j < 8; ++j) {
                const int q = j * 256 + tid;
                tr[j] = *(const uint4*)CHUNK_SRC(q, nextCol);
            }
        }
#endif

        // ---- compute: 16x16 tile per wave from shared B tile ---------------
        v8f accHH = {}; v8f accHL = {}; v8f accLH = {};
#pragma unroll
        for (int f = 0; f < 8; ++f) {
            const __bf16* pbh = &sB[buf][ss][0][rowIn][halfSel * 8 + f * 32];
            const __bf16* pbl = &sB[buf][ss][1][rowIn][halfSel * 8 + f * 32];
            v8bf b0 = *(const v8bf*)pbh;
            v8bf b1 = *(const v8bf*)(pbh + 16);
            v8bf c0 = *(const v8bf*)pbl;
            v8bf c1 = *(const v8bf*)(pbl + 16);
            v16bf bh, bl;
#pragma unroll
            for (int e = 0; e < 8; ++e) {
                bh[e] = b0[e]; bh[8 + e] = b1[e];
                bl[e] = c0[e]; bl[8 + e] = c1[e];
            }
            accHH = __builtin_amdgcn_wmma_f32_16x16x32_bf16(false, aHi[f], false, bh,
                                                            (short)0, accHH, false, false);
            accHL = __builtin_amdgcn_wmma_f32_16x16x32_bf16(false, aHi[f], false, bl,
                                                            (short)0, accHL, false, false);
            accLH = __builtin_amdgcn_wmma_f32_16x16x32_bf16(false, aLo[f], false, bh,
                                                            (short)0, accLH, false, false);
        }
        const int col = (it << 5) + (ss << 4) + rowIn;
        const float en = enorm[col];
#pragma unroll
        for (int v = 0; v < 8; ++v) {
            float dot = accHH[v] + accHL[v] + accLH[v];
            float d = en - 2.0f * dot;               // ||x||^2 constant per row
            if (d < runMin[v]) { runMin[v] = d; runIdx[v] = col; }
        }

#if USE_ASYNC
        WAIT_ASYNC();
        __syncthreads();
#else
        if (more) {
#pragma unroll
            for (int j = 0; j < 8; ++j) {
                const int q = j * 256 + tid;
                *(uint4*)CHUNK_DST(q, buf ^ 1) = tr[j];
            }
        }
        __syncthreads();
#endif
    }
#undef CHUNK_SRC
#undef CHUNK_DST

    // ---- argmin reduce: combine stripe-pair waves per row-block ------------
#pragma unroll
    for (int v = 0; v < 8; ++v) { sMin[wave][lane][v] = runMin[v]; sIdx[wave][lane][v] = runIdx[v]; }
    __syncthreads();
    if (tid < 64) {
        const int r    = tid;            // row within block's 64 rows
        const int rrb  = r >> 4;
        const int rIn  = r & 15;
        const int lbase = (rIn < 8) ? 0 : 16;
        const int slot  = rIn & 7;
        float best = 3.0e38f; int bi = 0;
#pragma unroll
        for (int p = 0; p < 2; ++p) {
            const int w = rrb + p * 4;
            for (int l = 0; l < 16; ++l) {
                float v = sMin[w][lbase + l][slot];
                if (v < best) { best = v; bi = sIdx[w][lbase + l][slot]; }
            }
        }
        idxOut[(blockIdx.x << 6) + r] = bi;
    }
}

// ---------------- epilogue kernels -------------------------------------------
__global__ void zero_enc(float4* __restrict__ p, long n4) {
    long i = (long)blockIdx.x * blockDim.x + threadIdx.x;
    const long stride = (long)gridDim.x * blockDim.x;
    float4 z; z.x = z.y = z.z = z.w = 0.0f;
    for (; i < n4; i += stride) p[i] = z;
}

__global__ void init_small(int* __restrict__ counts, float* __restrict__ lossacc) {
    int i = blockIdx.x * blockDim.x + threadIdx.x;
    if (i < K_CODES) counts[i] = 0;
    if (i == 0) *lossacc = 0.0f;
}

__global__ void scatter_enc(const int* __restrict__ idx, float* __restrict__ enc,
                            int* __restrict__ counts) {
    const int n = blockIdx.x * blockDim.x + threadIdx.x;   // N_VEC threads
    const int k = idx[n];
    enc[(size_t)n * K_CODES + k] = 1.0f;
    atomicAdd(&counts[k], 1);
}

__global__ void quant_loss(const float* __restrict__ in, const float* __restrict__ emb,
                           const int* __restrict__ idx, float* __restrict__ qout,
                           float* __restrict__ lossacc) {
    const int n = blockIdx.x, c = threadIdx.x;
    const int b = n >> 10, hw = n & 1023;
    const size_t off = (size_t)b * 262144 + (size_t)c * 1024 + hw;
    const int k = idx[n];
    const float q = emb[(size_t)k * C_DIM + c];
    const float x = in[off];
    qout[off] = q;                                   // straight-through == q
    const float d = q - x;
    __shared__ float red[256];
    red[c] = d * d;
    __syncthreads();
    for (int s = 128; s > 0; s >>= 1) { if (c < s) red[c] += red[c + s]; __syncthreads(); }
    if (c == 0) atomicAdd(lossacc, red[0]);
}

__global__ void finalize(const float* __restrict__ lossacc, const int* __restrict__ counts,
                         float* __restrict__ out_loss, float* __restrict__ out_perp) {
    __shared__ float red[256];
    const int t = threadIdx.x;
    float s = 0.0f;
    for (int k = t; k < K_CODES; k += 256) {
        float p = (float)counts[k] * (1.0f / (float)N_VEC);
        s += p * logf(p + 1e-10f);
    }
    red[t] = s;
    __syncthreads();
    for (int st = 128; st > 0; st >>= 1) { if (t < st) red[t] += red[t + st]; __syncthreads(); }
    if (t == 0) {
        *out_perp = expf(-red[0]);
        *out_loss = 0.25f * (*lossacc) / (float)((size_t)N_VEC * C_DIM);
    }
}

// ---------------- launcher ---------------------------------------------------
extern "C" void kernel_launch(void* const* d_in, const int* in_sizes, int n_in,
                              void* d_out, int out_size, void* d_ws, size_t ws_size,
                              hipStream_t stream) {
    const float* inputs = (const float*)d_in[0];   // [8,256,32,32]
    const float* emb    = (const float*)d_in[1];   // [8192,256]
    float* out = (float*)d_out;
    // out layout: [loss(1)] [quantized(2097152)] [perplexity(1)] [encodings(67108864)]
    float* out_loss = out;
    float* out_q    = out + 1;
    float* out_perp = out + 1 + 2097152;
    float* out_enc  = out + 2 + 2097152;

    char* ws = (char*)d_ws;
    __bf16* ehi   = (__bf16*)(ws);                      // 4 MB
    __bf16* elo   = (__bf16*)(ws + 4194304);            // 4 MB
    __bf16* xhi   = (__bf16*)(ws + 8388608);            // 4 MB
    __bf16* xlo   = (__bf16*)(ws + 12582912);           // 4 MB
    float*  enorm = (float*)(ws + 16777216);            // 32 KB
    int*    idx   = (int*)(ws + 16809984);              // 32 KB
    int*    counts= (int*)(ws + 16842752);              // 32 KB
    float*  lossa = (float*)(ws + 16875520);            // 4 B

    prep_embed <<<K_CODES, 256, 0, stream>>>(emb, ehi, elo, enorm);
    prep_inputs<<<N_VEC,   256, 0, stream>>>(inputs, xhi, xlo);
    init_small <<<(K_CODES + 255) / 256, 256, 0, stream>>>(counts, lossa);
    zero_enc   <<<8192, 256, 0, stream>>>((float4*)out_enc, (long)N_VEC * K_CODES / 4);
    vq_argmin  <<<N_VEC / 64, 256, 0, stream>>>(xhi, xlo, ehi, elo, enorm, idx);
    scatter_enc<<<N_VEC / 256, 256, 0, stream>>>(idx, out_enc, counts);
    quant_loss <<<N_VEC, 256, 0, stream>>>(inputs, emb, idx, out_q, lossa);
    finalize   <<<1, 256, 0, stream>>>(lossa, counts, out_loss, out_perp);
}
